// log_Sinkhorn_Distance_Loss_34591666602846
// MI455X (gfx1250) — matrix operand order repaired and checked by reading, not compiled, not run on hardware
//
#include <hip/hip_runtime.h>
#include <math.h>

typedef __attribute__((ext_vector_type(16))) _Float16 v16h;
typedef __attribute__((ext_vector_type(8)))  _Float16 v8h;
typedef __attribute__((ext_vector_type(8)))  float    v8f;

#define BATCH   16
#define NPT     2048
#define EPSR    0.1f
#define INV_EPS 10.0f
#define NITER   50

// ---------------------------------------------------------------------------
// Setup: build f16 hi/lo split row-vectors for the WMMA dot products,
// squared norms, zero the potentials.
//   dot(x,y) ~= xh.yh + xh.yl + xl.yh   (drop lo*lo, ~2^-22 relative)
//   A-row (x): [xh0 xh1 xh2 xh0 xh1 xh2 xl0 xl1 xl2 0...]   (K = 0..8)
//   B-row (y): [yh0 yh1 yh2 yl0 yl1 yl2 yh0 yh1 yh2 0...]
// ---------------------------------------------------------------------------
__global__ void setup_kernel(const float* __restrict__ x, const float* __restrict__ y,
                             _Float16* __restrict__ rvA, _Float16* __restrict__ rvB,
                             float* __restrict__ xsq, float* __restrict__ ysq,
                             float* __restrict__ u,   float* __restrict__ v)
{
    int t = blockIdx.x * blockDim.x + threadIdx.x;     // 0 .. B*N-1
    if (t >= BATCH * NPT) return;
    const float* xp = x + (size_t)t * 3;
    const float* yp = y + (size_t)t * 3;
    float xs = 0.f, ys = 0.f;
    _Float16 a16[16], b16[16];
    #pragma unroll
    for (int k = 0; k < 16; ++k) { a16[k] = (_Float16)0.f; b16[k] = (_Float16)0.f; }
    #pragma unroll
    for (int d = 0; d < 3; ++d) {
        float xv = xp[d], yv = yp[d];
        xs += xv * xv; ys += yv * yv;
        _Float16 xh = (_Float16)xv; _Float16 xl = (_Float16)(xv - (float)xh);
        _Float16 yh = (_Float16)yv; _Float16 yl = (_Float16)(yv - (float)yh);
        a16[d] = xh; a16[3 + d] = xh; a16[6 + d] = xl;
        b16[d] = yh; b16[3 + d] = yl; b16[6 + d] = yh;
    }
    _Float16* ra = rvA + (size_t)t * 16;
    _Float16* rb = rvB + (size_t)t * 16;
    #pragma unroll
    for (int k = 0; k < 16; ++k) { ra[k] = a16[k]; rb[k] = b16[k]; }
    xsq[t] = xs; ysq[t] = ys;
    u[t] = 0.f; v[t] = 0.f;
}

// ---------------------------------------------------------------------------
// One Sinkhorn half-iteration: for each row i of the "t" side,
//   t_i <- eps*(log_marg - LSE_j[(-C_ij + t_i + s_j)/eps]) + t_i
// Called with (rvA,rvB,xsq,ysq,u,v) for the u-update and with the two sides
// swapped (rvB,rvA,ysq,xsq,v,u) for the v-update (C is symmetric in roles).
// grid = (16 row-groups, 16 batches), block = 256 (8 waves), wave owns 16 rows.
// ---------------------------------------------------------------------------
__global__ __launch_bounds__(256) void sink_pass(
    const _Float16* __restrict__ rvT, const _Float16* __restrict__ rvS,
    const float* __restrict__ tsq,    const float* __restrict__ ssq,
    float* __restrict__ tpot,         const float* __restrict__ spot,
    float log_marg)
{
    __shared__ float sh_s[NPT];
    __shared__ float sh_q[NPT];
    const int b    = blockIdx.y;
    const int rg   = blockIdx.x;
    const int tid  = threadIdx.x;
    const int wave = tid >> 5;
    const int lane = tid & 31;
    const int half = lane >> 4;          // 0: lanes 0-15, 1: lanes 16-31
    const int lp   = lane & 15;
    const size_t bo = (size_t)b * NPT;

    for (int idx = tid; idx < NPT; idx += 256) {
        sh_s[idx] = spot[bo + idx];
        sh_q[idx] = ssq[bo + idx];
    }
    __syncthreads();

    const int i0 = rg * 128 + wave * 16;
    v8h zero8 = {};

    // A fragment (16x32 f16): lanes 0-15 carry K0-7, lanes 16-31 carry K8-15
    // in halves 0..7; halves 8..15 (K16..31) are zero.
    union Frag { v16h v; v8h h[2]; } af, bf;
    {
        const _Float16* aptr = rvT + (bo + (size_t)(i0 + lp)) * 16 + (half ? 8 : 0);
        af.h[0] = *(const v8h*)aptr;
        af.h[1] = zero8;
    }

    // Per-slot row constants: slot r -> row i0 + r + 8*half.
    float usq[8], uin[8], mx[8], sm[8];
    #pragma unroll
    for (int r = 0; r < 8; ++r) {
        int i = i0 + r + 8 * half;
        usq[r] = tsq[bo + i];
        uin[r] = tpot[bo + i];
        mx[r] = -__builtin_inff();
        sm[r] = 0.f;
    }

    #pragma unroll 2
    for (int jc = 0; jc < NPT; jc += 16) {
        // B fragment (32x16 f16): column jc+lp; lanes 0-15 carry K0-7, lanes
        // 16-31 carry K8-15 in halves 0..7; halves 8..15 (K16..31) zero.
        {
            const _Float16* bptr = rvS + (bo + (size_t)(jc + lp)) * 16 + (half ? 8 : 0);
            bf.h[0] = *(const v8h*)bptr;
            bf.h[1] = zero8;
        }
        v8f cz = {};
        v8f dots = __builtin_amdgcn_wmma_f32_16x16x32_f16(
            false, af.v, false, bf.v, (short)0, cz, false, false);

        float sj = sh_s[jc + lp];
        float qj = sh_q[jc + lp];
        #pragma unroll
        for (int r = 0; r < 8; ++r) {
            float c2  = usq[r] + qj - 2.0f * dots[r];
            // hardware v_sqrt_f32: ~1ulp in the normal range, plenty here
            float c   = __builtin_amdgcn_sqrtf(fmaxf(c2, 0.f));
            float val = (uin[r] + sj - c) * INV_EPS;
            // branchless online logsumexp (single exp per element)
            float nm = fmaxf(mx[r], val);
            float e  = __expf(fminf(mx[r], val) - nm);   // exp(-|diff|); exp(-inf)=0
            bool  gt = val > mx[r];
            sm[r] = gt ? (sm[r] * e + 1.0f) : (sm[r] + e);
            mx[r] = nm;
        }
    }

    // Merge the 16 strided partial LSEs across each 16-lane group.
    #pragma unroll
    for (int r = 0; r < 8; ++r) {
        float m = mx[r], s = sm[r];
        #pragma unroll
        for (int d = 1; d < 16; d <<= 1) {
            float om = __shfl_xor(m, d, 32);
            float os = __shfl_xor(s, d, 32);
            float nm = fmaxf(m, om);
            s = s * __expf(m - nm) + os * __expf(om - nm);
            m = nm;
        }
        float lse  = m + __logf(s);
        float tnew = EPSR * (log_marg - lse) + uin[r];
        if (lp == r) tpot[bo + (size_t)(i0 + r + 8 * half)] = tnew;
    }
}

// ---------------------------------------------------------------------------
// Final: P = exp((-C+u_i+v_j)/eps), write C and P, per-wave partial of
// sum(P*C) for a deterministic two-stage cost reduction.
// ---------------------------------------------------------------------------
__global__ __launch_bounds__(256) void sink_final(
    const _Float16* __restrict__ rvA, const _Float16* __restrict__ rvB,
    const float* __restrict__ xsq,    const float* __restrict__ ysq,
    const float* __restrict__ u,      const float* __restrict__ v,
    float* __restrict__ P, float* __restrict__ C, float* __restrict__ partials)
{
    __shared__ float sh_v[NPT];
    __shared__ float sh_q[NPT];
    const int b    = blockIdx.y;
    const int rg   = blockIdx.x;
    const int tid  = threadIdx.x;
    const int wave = tid >> 5;
    const int lane = tid & 31;
    const int half = lane >> 4;
    const int lp   = lane & 15;
    const size_t bo = (size_t)b * NPT;

    for (int idx = tid; idx < NPT; idx += 256) {
        sh_v[idx] = v[bo + idx];
        sh_q[idx] = ysq[bo + idx];
    }
    __syncthreads();

    const int i0 = rg * 128 + wave * 16;
    v8h zero8 = {};
    union Frag { v16h v; v8h h[2]; } af, bf;
    {
        const _Float16* aptr = rvA + (bo + (size_t)(i0 + lp)) * 16 + (half ? 8 : 0);
        af.h[0] = *(const v8h*)aptr;
        af.h[1] = zero8;
    }

    float usq[8], uin[8];
    #pragma unroll
    for (int r = 0; r < 8; ++r) {
        int i = i0 + r + 8 * half;
        usq[r] = xsq[bo + i];
        uin[r] = u[bo + i];
    }

    float acc = 0.f;
    #pragma unroll 2
    for (int jc = 0; jc < NPT; jc += 16) {
        {
            const _Float16* bptr = rvB + (bo + (size_t)(jc + lp)) * 16 + (half ? 8 : 0);
            bf.h[0] = *(const v8h*)bptr;
            bf.h[1] = zero8;
        }
        v8f cz = {};
        v8f dots = __builtin_amdgcn_wmma_f32_16x16x32_f16(
            false, af.v, false, bf.v, (short)0, cz, false, false);

        float vj = sh_v[jc + lp];
        float qj = sh_q[jc + lp];
        #pragma unroll
        for (int r = 0; r < 8; ++r) {
            int   i  = i0 + r + 8 * half;
            float c2 = usq[r] + qj - 2.0f * dots[r];
            float c  = __builtin_amdgcn_sqrtf(fmaxf(c2, 0.f));
            float p  = __expf((uin[r] + vj - c) * INV_EPS);
            size_t off = (bo + (size_t)i) * NPT + (size_t)(jc + lp);
            C[off] = c;
            P[off] = p;
            acc += p * c;
        }
    }

    #pragma unroll
    for (int d = 16; d >= 1; d >>= 1) acc += __shfl_xor(acc, d, 32);
    if (lane == 0) partials[((size_t)b * 16 + rg) * 8 + wave] = acc;
}

__global__ void reduce_cost(const float* __restrict__ partials, float* __restrict__ cost)
{
    __shared__ float sh[128];
    int b = blockIdx.x, t = threadIdx.x;
    sh[t] = partials[(size_t)b * 128 + t];
    __syncthreads();
    for (int s = 64; s > 0; s >>= 1) {
        if (t < s) sh[t] += sh[t + s];
        __syncthreads();
    }
    if (t == 0) cost[b] = sh[0];
}

// ---------------------------------------------------------------------------
extern "C" void kernel_launch(void* const* d_in, const int* in_sizes, int n_in,
                              void* d_out, int out_size, void* d_ws, size_t ws_size,
                              hipStream_t stream)
{
    (void)in_sizes; (void)n_in; (void)out_size; (void)ws_size;
    const float* x = (const float*)d_in[0];   // [16,2048,3] f32
    const float* y = (const float*)d_in[1];   // [16,2048,3] f32

    float* out  = (float*)d_out;
    float* cost = out;                                  // [16]
    float* P    = out + 16;                             // [16,2048,2048]
    float* C    = P + (size_t)BATCH * NPT * NPT;        // [16,2048,2048]

    // workspace layout
    char* ws = (char*)d_ws;
    float*    u        = (float*)(ws);                         // 128 KB
    float*    v        = (float*)(ws + 131072);                // 128 KB
    float*    xsq      = (float*)(ws + 262144);                // 128 KB
    float*    ysq      = (float*)(ws + 393216);                // 128 KB
    _Float16* rvA      = (_Float16*)(ws + 524288);             // 1 MB
    _Float16* rvB      = (_Float16*)(ws + 524288 + 1048576);   // 1 MB
    float*    partials = (float*)(ws + 524288 + 2 * 1048576);  // 8 KB

    float log_marg = logf(1.0f / (float)NPT + 1e-8f);

    setup_kernel<<<dim3((BATCH * NPT + 255) / 256), dim3(256), 0, stream>>>(
        x, y, rvA, rvB, xsq, ysq, u, v);

    dim3 grid(16, BATCH), blk(256);
    for (int it = 0; it < NITER; ++it) {
        sink_pass<<<grid, blk, 0, stream>>>(rvA, rvB, xsq, ysq, u, v, log_marg);
        sink_pass<<<grid, blk, 0, stream>>>(rvB, rvA, ysq, xsq, v, u, log_marg);
    }

    sink_final<<<grid, blk, 0, stream>>>(rvA, rvB, xsq, ysq, u, v, P, C, partials);
    reduce_cost<<<dim3(16), dim3(128), 0, stream>>>(partials, cost);
}
